// ModelToVolumeAligner_35613868818597
// MI455X (gfx1250) — matrix operand order repaired and checked by reading, not compiled, not run on hardware
//
#include <hip/hip_runtime.h>
#include <hip/hip_bf16.h>
#include <math.h>

typedef __attribute__((ext_vector_type(16))) _Float16 v16h;
typedef __attribute__((ext_vector_type(8)))  _Float16 v8h;
typedef __attribute__((ext_vector_type(8)))  float    v8f;

#define KT       102400          // 20480 atoms * 5 gaussians
#define NA       20480
#define NG       128             // grid edge
#define HALF_NG  64
#define KSTEP    256             // K per double-buffered stage (8 WMMAs)

// ---- static device storage (graph-capture safe; fully overwritten each call) ----
__device__ __attribute__((aligned(256))) float    g_px[KT];
__device__ __attribute__((aligned(256))) float    g_py[KT];
__device__ __attribute__((aligned(256))) float    g_pz[KT];
__device__ __attribute__((aligned(256))) float    g_nc[KT];    // -0.5/var
__device__ __attribute__((aligned(256))) float    g_pref[KT];  // amp*(2*pi*var)^-1.5
// k-major 1-D gaussian tables: [coord][k], f16  (~26 MB each, all three fit in L2)
__device__ __attribute__((aligned(256))) _Float16 g_gx[NG * KT];
__device__ __attribute__((aligned(256))) _Float16 g_gy[NG * KT];
__device__ __attribute__((aligned(256))) _Float16 g_gz[NG * KT]; // prefactor folded in

// ---------------------------------------------------------------------------
// 1) rotate atoms, expand per-(atom,gaussian) constants
// ---------------------------------------------------------------------------
__global__ __launch_bounds__(256) void prep_kernel(
    const float* __restrict__ quat, const float* __restrict__ offset,
    const float* __restrict__ positions, const float* __restrict__ amps,
    const float* __restrict__ vars)
{
    int k = blockIdx.x * 256 + threadIdx.x;
    if (k >= KT) return;

    float q0 = quat[0], q1 = quat[1], q2 = quat[2], q3 = quat[3];
    float inv = rsqrtf(q0 * q0 + q1 * q1 + q2 * q2 + q3 * q3);
    float w = q0 * inv, x = q1 * inv, y = q2 * inv, z = q3 * inv;
    float r00 = 1.f - 2.f * (y * y + z * z), r01 = 2.f * (x * y - w * z), r02 = 2.f * (x * z + w * y);
    float r10 = 2.f * (x * y + w * z), r11 = 1.f - 2.f * (x * x + z * z), r12 = 2.f * (y * z - w * x);
    float r20 = 2.f * (x * z - w * y), r21 = 2.f * (y * z + w * x), r22 = 1.f - 2.f * (x * x + y * y);

    int a = k / 5;
    float p0 = positions[a * 3 + 0], p1 = positions[a * 3 + 1], p2 = positions[a * 3 + 2];
    g_px[k] = p0 * r00 + p1 * r10 + p2 * r20 + offset[0];
    g_py[k] = p0 * r01 + p1 * r11 + p2 * r21 + offset[1];
    g_pz[k] = p0 * r02 + p1 * r12 + p2 * r22 + offset[2];

    float v   = vars[k];
    float amp = amps[k];
    float t   = 6.283185307179586f * v;
    g_pref[k] = amp / (t * sqrtf(t));   // (2*pi*var)^-1.5 * amp
    g_nc[k]   = -0.5f / v;
}

// ---------------------------------------------------------------------------
// 2) build k-major 1-D gaussian tables (f16)
// ---------------------------------------------------------------------------
__global__ __launch_bounds__(256) void table_kernel()
{
    int k = blockIdx.x * 256 + threadIdx.x;   // 400 blocks * 256 = 102400
    int c = blockIdx.y;                       // 0..127
    float fc = (float)(c - HALF_NG);
    float nc = g_nc[k];
    float dx = fc - g_px[k];
    float dy = fc - g_py[k];
    float dz = fc - g_pz[k];
    size_t idx = (size_t)c * KT + k;
    g_gx[idx] = (_Float16)expf(nc * dx * dx);
    g_gy[idx] = (_Float16)expf(nc * dy * dy);
    g_gz[idx] = (_Float16)(g_pref[k] * expf(nc * dz * dz));
}

// ---------------------------------------------------------------------------
// 3) WMMA GEMM splat + fused correlation partials
//    block = (z, y-block of 16); 8 waves, each owns 16 x-columns
//    vol[(z,y), x] = sum_k (gz[k,z]*gy[k,y]) * gx[k,x]
//    Double-buffered A tile in LDS (1 barrier / 8 WMMAs), dual accumulators
//    to break the WMMA C->D serial dependency chain.
// ---------------------------------------------------------------------------
__global__ __launch_bounds__(256) void gemm_splat_kernel(
    const float* __restrict__ grid, float* __restrict__ part)
{
    __shared__ __attribute__((aligned(128))) _Float16 ldsA[2][16 * KSTEP]; // A[m][k]
    __shared__ float red[256];

    int bid  = blockIdx.x;
    int z    = bid >> 3;
    int y0   = (bid & 7) << 4;
    int tid  = threadIdx.x;
    int lane = tid & 31;
    int wave = tid >> 5;
    int x0   = wave << 4;

    const _Float16* __restrict__ gzrow = g_gz + (size_t)z * KT;

    // build-phase mapping: thread t owns row m = t>>4, 8 contiguous k at k8,
    // for both k-halves of the 256-wide stage -> v8h loads, v_pk_mul_f16,
    // ds_store_b128 (fully vectorized, fully coalesced).
    int bm = tid >> 4;
    int k8 = (tid & 15) << 3;
    const _Float16* __restrict__ gyrow = g_gy + (size_t)(y0 + bm) * KT + k8;

    v8f acc0 = {0.f, 0.f, 0.f, 0.f, 0.f, 0.f, 0.f, 0.f};
    v8f acc1 = {0.f, 0.f, 0.f, 0.f, 0.f, 0.f, 0.f, 0.f};

    int mrow  = lane & 15;                  // A: M row / B: N column
    int abase = (lane & 16) ? 16 : 0;       // per-lane K base for A frag
    int bbase = (lane & 16) ? 16 : 0;       // per-lane K base for B frag
    const _Float16* __restrict__ browK = g_gx + (size_t)(x0 + mrow) * KT + bbase;

    // prologue: build stage 0 into buffer 0
    {
        v8h gz0 = *(const v8h*)(gzrow + k8);
        v8h gz1 = *(const v8h*)(gzrow + 128 + k8);
        v8h gy0 = *(const v8h*)(gyrow);
        v8h gy1 = *(const v8h*)(gyrow + 128);
        *(v8h*)(&ldsA[0][bm * KSTEP + k8])       = gz0 * gy0;
        *(v8h*)(&ldsA[0][bm * KSTEP + 128 + k8]) = gz1 * gy1;
    }

    int cur = 0;
    for (int kc = 0; kc < KT; kc += KSTEP) {
        __syncthreads();   // buffer `cur` ready; buffer `cur^1` free

        // build next stage into the other buffer (overlaps with WMMAs below)
        if (kc + KSTEP < KT) {
            int kn = kc + KSTEP;
            v8h gz0 = *(const v8h*)(gzrow + kn + k8);
            v8h gz1 = *(const v8h*)(gzrow + kn + 128 + k8);
            v8h gy0 = *(const v8h*)(gyrow + kn);
            v8h gy1 = *(const v8h*)(gyrow + kn + 128);
            *(v8h*)(&ldsA[cur ^ 1][bm * KSTEP + k8])       = gz0 * gy0;
            *(v8h*)(&ldsA[cur ^ 1][bm * KSTEP + 128 + k8]) = gz1 * gy1;
        }

        const _Float16* arow = &ldsA[cur][mrow * KSTEP + abase];
        #pragma unroll
        for (int s = 0; s < 8; s += 2) {
            // A frag: K = abase+0..7 (VGPR0-3) and abase+8..15 (VGPR4-7)
            v8h alo0 = *(const v8h*)(arow + s * 32);
            v8h ahi0 = *(const v8h*)(arow + s * 32 + 8);
            v16h a0  = __builtin_shufflevector(alo0, ahi0,
                         0, 1, 2, 3, 4, 5, 6, 7, 8, 9, 10, 11, 12, 13, 14, 15);
            v16h b0  = *(const v16h*)(browK + kc + s * 32);

            v8h alo1 = *(const v8h*)(arow + (s + 1) * 32);
            v8h ahi1 = *(const v8h*)(arow + (s + 1) * 32 + 8);
            v16h a1  = __builtin_shufflevector(alo1, ahi1,
                         0, 1, 2, 3, 4, 5, 6, 7, 8, 9, 10, 11, 12, 13, 14, 15);
            v16h b1  = *(const v16h*)(browK + kc + (s + 1) * 32);

            // independent accumulators -> adjacent WMMAs have no RAW dependency
            acc0 = __builtin_amdgcn_wmma_f32_16x16x32_f16(
                       false, a0, false, b0, (short)0, acc0, false, false);
            acc1 = __builtin_amdgcn_wmma_f32_16x16x32_f16(
                       false, a1, false, b1, (short)0, acc1, false, false);
        }
        cur ^= 1;
    }

    v8f acc = acc0 + acc1;

    // fused correlation partials: C layout VGPR r -> M = r + (lane<16?0:8), N = lane%16
    int n    = lane & 15;
    int moff = (lane & 16) ? 8 : 0;
    const float* gbase = grid + ((size_t)z * NG + (y0 + moff)) * NG + x0 + n;
    float svg = 0.f, svv = 0.f;
    #pragma unroll
    for (int r = 0; r < 8; ++r) {
        float v = acc[r];
        float g = gbase[(size_t)r * NG];
        svg += v * g;
        svv += v * v;
    }

    red[tid] = svg; __syncthreads();
    for (int s = 128; s; s >>= 1) { if (tid < s) red[tid] += red[tid + s]; __syncthreads(); }
    if (tid == 0) part[bid * 2 + 0] = red[0];
    __syncthreads();
    red[tid] = svv; __syncthreads();
    for (int s = 128; s; s >>= 1) { if (tid < s) red[tid] += red[tid + s]; __syncthreads(); }
    if (tid == 0) part[bid * 2 + 1] = red[0];
}

// ---------------------------------------------------------------------------
// 4) deterministic single-block finalize: 1 - sum(v*g)/(||v||*||g||)
//    (the reference's v/v.sum() cancels in the correlation)
// ---------------------------------------------------------------------------
__global__ __launch_bounds__(256) void finalize_kernel(
    const float* __restrict__ grid, const float* __restrict__ part,
    float* __restrict__ out)
{
    __shared__ float r0[256], r1[256], r2[256];
    int tid = threadIdx.x;
    float svg = 0.f, svv = 0.f, sgg = 0.f;
    for (int i = tid; i < 1024; i += 256) {
        svg += part[i * 2 + 0];
        svv += part[i * 2 + 1];
    }
    const int NVOX = NG * NG * NG;
    for (int i = tid; i < NVOX; i += 256) {
        float g = grid[i];
        sgg += g * g;
    }
    r0[tid] = svg; r1[tid] = svv; r2[tid] = sgg; __syncthreads();
    for (int s = 128; s; s >>= 1) {
        if (tid < s) { r0[tid] += r0[tid + s]; r1[tid] += r1[tid + s]; r2[tid] += r2[tid + s]; }
        __syncthreads();
    }
    if (tid == 0)
        out[0] = 1.0f - r0[0] / (sqrtf(r1[0]) * sqrtf(r2[0]));
}

// ---------------------------------------------------------------------------
extern "C" void kernel_launch(void* const* d_in, const int* in_sizes, int n_in,
                              void* d_out, int out_size, void* d_ws, size_t ws_size,
                              hipStream_t stream)
{
    const float* quat      = (const float*)d_in[0];
    const float* offset    = (const float*)d_in[1];
    const float* positions = (const float*)d_in[2];
    const float* amps      = (const float*)d_in[3];
    const float* vars      = (const float*)d_in[4];
    const float* grid      = (const float*)d_in[5];
    float*       out       = (float*)d_out;
    float*       part      = (float*)d_ws;   // 1024*2 floats = 8 KB

    prep_kernel<<<KT / 256, 256, 0, stream>>>(quat, offset, positions, amps, vars);
    table_kernel<<<dim3(KT / 256, NG), 256, 0, stream>>>();
    gemm_splat_kernel<<<NG * 8, 256, 0, stream>>>(grid, part);
    finalize_kernel<<<1, 256, 0, stream>>>(grid, part, out);
}
